// FFTv2_19490561589939
// MI455X (gfx1250) — matrix operand order repaired and checked by reading, not compiled, not run on hardware
//
#include <hip/hip_runtime.h>
#include <cmath>

// ---------------------------------------------------------------------------
// CDNA5 (gfx1250) fused decimated-FFT kernel.
//   Leaf 128x128 DFT  -> v_wmma_f32_16x16x32_bf16 (split-precision bf16)
//   3 combine stages  -> in-place fp32 butterflies in LDS
// ---------------------------------------------------------------------------

typedef __attribute__((ext_vector_type(16))) __bf16          v16bf;
typedef __attribute__((ext_vector_type(16))) unsigned short  us16;
typedef __attribute__((ext_vector_type(8)))  unsigned int    v8u;
typedef __attribute__((ext_vector_type(8)))  float           v8f;

#define N_FFT   1024
#define LEAF    128
#define BATCH   32768
#define TILE_B  16

__device__ __forceinline__ unsigned short bf16rn(float f) {
    unsigned u = __builtin_bit_cast(unsigned, f);
    return (unsigned short)((u + 0x7FFFu + ((u >> 16) & 1u)) >> 16);  // RNE
}

__device__ __forceinline__ v8f wmma_bf16(v16bf a, v16bf b, v8f c) {
    // v_wmma_f32_16x16x32_bf16  D = A*B + C
    return __builtin_amdgcn_wmma_f32_16x16x32_bf16(
        /*neg_a=*/false, a, /*neg_b=*/false, b,
        /*c_mod=*/(short)0, c, /*reuse_a=*/false, /*reuse_b=*/false);
}

// Build a 16x32 bf16 A-fragment from LDS.  Contiguous-K order (matches frag_b): lane holds row
// M = lane&15 and one contiguous 16-element K run (K = kb..kb+15).
// p points at the run (16B aligned); second half is +8 u16 (= +16B).
__device__ __forceinline__ v16bf frag_a(const unsigned short* p) {
    const uint4* q = (const uint4*)p;
    uint4 a = q[0];      // K run 0 (8 x bf16)
    uint4 b = q[1];      // K run 1 (8 x bf16), +16 bytes
    v8u t;
    t[0] = a.x; t[1] = a.y; t[2] = a.z; t[3] = a.w;
    t[4] = b.x; t[5] = b.y; t[6] = b.z; t[7] = b.w;
    return __builtin_bit_cast(v16bf, t);
}

// Build a 32x16 bf16 B-fragment from a global f32 DFT matrix.  Contiguous-K order (matches frag_a):
// lane holds column N = n0+(lane&15) and 16 contiguous K rows starting at
// col = kc + (lane>>4)*16.  The DFT matrix is symmetric, so column reads
// become contiguous row reads (coalescable float4s).
__device__ __forceinline__ v16bf frag_b(const float* __restrict__ M, int row, int col) {
    const float* p = M + row * LEAF + col;
    us16 u;
#pragma unroll
    for (int i = 0; i < 16; i += 4) {
        float4 f = *(const float4*)(p + i);
        u[i + 0] = bf16rn(f.x);
        u[i + 1] = bf16rn(f.y);
        u[i + 2] = bf16rn(f.z);
        u[i + 3] = bf16rn(f.w);
    }
    return __builtin_bit_cast(v16bf, u);
}

__global__ __launch_bounds__(256)
void fftv2_wmma_kernel(const float* __restrict__ x,
                       const float* __restrict__ fr_u, const float* __restrict__ fr_l,
                       const float* __restrict__ fi_u, const float* __restrict__ fi_l,
                       float* __restrict__ outR, float* __restrict__ outI) {
    // LDS union region:
    //   phase 1: xs_hi (17408 u16) + xs_lo (17408 u16) = 69632 B  (padded rows: 136 u16)
    //   phase 2: res_r (16x1032 f32) + res_i            = 132096 B (padded rows: 1032 f32)
    __shared__ float s_u[33024];     // 132096 B
    __shared__ float s_tw[1792];     // 896 twiddles (cos,sin) = 7168 B

    const int tid     = threadIdx.x;
    const int lane    = tid & 31;                 // wave32
    const int wv      = tid >> 5;                 // 8 waves / block
    const int rowbase = blockIdx.x * TILE_B;

    unsigned short* xh = (unsigned short*)s_u;    // bf16-hi plane
    unsigned short* xl = xh + 17408;              // bf16-lo plane
    float* rr = s_u;                              // real results (phase 2)
    float* ri = s_u + 16512;                      // imag results (phase 2)

    const int ln = lane & 15;
    const int kk = lane >> 4;
    const int n0 = wv * 16;                       // this wave's N-tile (output k columns)

    // ---- B fragments (register resident, reused across all 8 groups) ----
    v16bf Brh[4], Brl[4], Bih[4], Bil[4];
    const int brow = n0 + ln;                     // symmetric matrix: column == row
#pragma unroll
    for (int c = 0; c < 4; ++c) {
        int col = c * 32 + kk * 16;
        Brh[c] = frag_b(fr_u, brow, col);
        Brl[c] = frag_b(fr_l, brow, col);
        Bih[c] = frag_b(fi_u, brow, col);
        Bil[c] = frag_b(fi_l, brow, col);
    }

    // ---- twiddle table (stage s: offset size-128, size = 128<<s) ----
    for (int u = tid; u < 896; u += 256) {
        int size, k;
        if      (u < 128) { size = 128; k = u;       }
        else if (u < 384) { size = 256; k = u - 128; }
        else              { size = 512; k = u - 384; }
        float th = -3.14159265358979323846f * (float)k / (float)size;
        s_tw[2 * u + 0] = cosf(th);
        s_tw[2 * u + 1] = sinf(th);
    }

    // ---- load x tile, split fp32 -> bf16 hi/lo, decimate into group planes ----
#pragma unroll
    for (int j = 0; j < 16; ++j) {
        int idx4 = tid + 256 * j;                 // 4096 float4s per tile
        int r    = idx4 >> 8;
        int c4   = idx4 & 255;
        float4 v = *(const float4*)(x + (size_t)(rowbase + r) * N_FFT + c4 * 4);
        float a4[4] = {v.x, v.y, v.z, v.w};
#pragma unroll
        for (int cc = 0; cc < 4; ++cc) {
            int c = c4 * 4 + cc;
            int n = c >> 3;
            int o = c & 7;
            int g = ((o & 1) << 2) | (o & 2) | ((o >> 2) & 1);   // bitrev3 (involution)
            float f  = a4[cc];
            unsigned ub = __builtin_bit_cast(unsigned, f);
            unsigned rh = (ub + 0x7FFFu + ((ub >> 16) & 1u)) >> 16;
            float hif   = __builtin_bit_cast(float, rh << 16);
            float lof   = f - hif;
            unsigned ul = __builtin_bit_cast(unsigned, lof);
            unsigned rl = (ul + 0x7FFFu + ((ul >> 16) & 1u)) >> 16;
            int off = (g * 16 + r) * 136 + n;     // padded row stride: 136 u16
            xh[off] = (unsigned short)rh;
            xl[off] = (unsigned short)rl;
        }
    }
    __syncthreads();

    // ---- leaf DFT via WMMA: (x_hi + x_lo) @ (F_u + F_l), real & imag ----
    v8f ar[8], ai[8];
#pragma unroll
    for (int g = 0; g < 8; ++g) { v8f z = {}; ar[g] = z; ai[g] = z; }

    const int m  = ln;        // A row (batch row in tile)
    const int kb = kk * 16;   // A K-run base within 32-chunk
#pragma unroll
    for (int g = 0; g < 8; ++g) {
#pragma unroll
        for (int c = 0; c < 4; ++c) {
            int base = (g * 16 + m) * 136 + c * 32 + kb;
            v16bf ahi = frag_a(xh + base);
            v16bf alo = frag_a(xl + base);
            ar[g] = wmma_bf16(ahi, Brh[c], ar[g]);
            ar[g] = wmma_bf16(ahi, Brl[c], ar[g]);
            ar[g] = wmma_bf16(alo, Brh[c], ar[g]);
            ar[g] = wmma_bf16(alo, Brl[c], ar[g]);
            ai[g] = wmma_bf16(ahi, Bih[c], ai[g]);
            ai[g] = wmma_bf16(ahi, Bil[c], ai[g]);
            ai[g] = wmma_bf16(alo, Bih[c], ai[g]);
            ai[g] = wmma_bf16(alo, Bil[c], ai[g]);
        }
    }
    __syncthreads();   // xs planes dead -> reuse region for results

    // ---- dump accumulators: res[b][g*128 + k], flat layout so that the
    //      radix-2 concat is an in-place butterfly ----
    {
        int b0 = kk * 8;
        int k  = n0 + ln;
#pragma unroll
        for (int g = 0; g < 8; ++g) {
#pragma unroll
            for (int r8 = 0; r8 < 8; ++r8) {
                int b = r8 + b0;                  // C layout: M = r8 + 8*(lane>>4)
                rr[b * 1032 + g * 128 + k] = ar[g][r8];
                ri[b * 1032 + g * 128 + k] = ai[g][r8];
            }
        }
    }
    __syncthreads();

    // ---- 3 radix-2 combine stages, in place in LDS ----
    for (int s = 0; s < 3; ++s) {
        int size  = 128 << s;
        int twoff = size - 128;
        for (int it = 0; it < 32; ++it) {
            int beta = tid + it * 256;            // 8192 butterflies per stage
            int b    = beta >> 9;
            int j    = beta & 511;
            int p    = j >> (7 + s);
            int k    = j & (size - 1);
            int base = b * 1032 + p * (size << 1) + k;
            float evr = rr[base],        odr = rr[base + size];
            float evi = ri[base],        odi = ri[base + size];
            float cw  = s_tw[2 * (twoff + k)];
            float sw  = s_tw[2 * (twoff + k) + 1];
            float tr  = cw * odr - sw * odi;
            float ti  = cw * odi + sw * odr;
            rr[base]        = evr + tr;
            rr[base + size] = evr - tr;
            ri[base]        = evi + ti;
            ri[base + size] = evi - ti;
        }
        __syncthreads();
    }

    // ---- coalesced float4 stores of real & imag ----
#pragma unroll
    for (int j = 0; j < 16; ++j) {
        int idx4 = tid + 256 * j;
        int r    = idx4 >> 8;
        int c4   = idx4 & 255;
        float4 vr = *(const float4*)&rr[r * 1032 + c4 * 4];
        float4 vi = *(const float4*)&ri[r * 1032 + c4 * 4];
        *(float4*)(outR + (size_t)(rowbase + r) * N_FFT + c4 * 4) = vr;
        *(float4*)(outI + (size_t)(rowbase + r) * N_FFT + c4 * 4) = vi;
    }
}

extern "C" void kernel_launch(void* const* d_in, const int* in_sizes, int n_in,
                              void* d_out, int out_size, void* d_ws, size_t ws_size,
                              hipStream_t stream) {
    (void)in_sizes; (void)n_in; (void)out_size; (void)d_ws; (void)ws_size;
    const float* x    = (const float*)d_in[0];
    const float* fr_u = (const float*)d_in[1];
    const float* fr_l = (const float*)d_in[2];
    const float* fi_u = (const float*)d_in[3];
    const float* fi_l = (const float*)d_in[4];
    float* outR = (float*)d_out;
    float* outI = outR + (size_t)BATCH * N_FFT;

    dim3 grid(BATCH / TILE_B);   // 2048 workgroups
    dim3 block(256);             // 8 wave32 waves
    fftv2_wmma_kernel<<<grid, block, 0, stream>>>(x, fr_u, fr_l, fi_u, fi_l, outR, outI);
}
